// MultiHeadAttention_82566451298956
// MI455X (gfx1250) — matrix operand order repaired and checked by reading, not compiled, not run on hardware
//
#include <hip/hip_runtime.h>
#include <hip/hip_bf16.h>

// ---------------------------------------------------------------------------
// MultiHeadAttention (causal), B=2 S=2048 D=1024 H=16 DH=64, fp32 in/out.
// fp16 WMMA (v_wmma_f32_16x16x32_f16) + flash attention (scores never
// materialized: 512MB > 192MB L2). K tiles staged into LDS by the Tensor
// Data Mover (tensor_load_to_lds, double buffered, s_wait_tensorcnt),
// projection GEMM register-double-buffered to avoid s_wait_loadcnt 0 stalls.
// ---------------------------------------------------------------------------

typedef __attribute__((ext_vector_type(16))) _Float16 v16h;
typedef __attribute__((ext_vector_type(8)))  float    v8f;
typedef __attribute__((ext_vector_type(4)))  unsigned int u32x4;
typedef __attribute__((ext_vector_type(4)))  unsigned int v4u;
typedef __attribute__((ext_vector_type(8)))  int      v8i;
typedef __attribute__((ext_vector_type(4)))  int      v4i;

union AFrag { v16h v; u32x4 q[2]; };   // 32 bytes: one 16x16x32 A or B fragment

static __device__ __forceinline__ v8f wmma_f16(v16h a, v16h b, v8f c) {
    // (neg_a, A, neg_b, B, c_mod, C, reuse_a, reuse_b)
    return __builtin_amdgcn_wmma_f32_16x16x32_f16(false, a, false, b,
                                                  (short)0, c, false, false);
}

#define NB  2
#define NS  2048
#define ND  1024
#define NH  16
#define NDH 64

#if defined(__has_builtin)
#if __has_builtin(__builtin_amdgcn_tensor_load_to_lds) && \
    __has_builtin(__builtin_amdgcn_s_wait_tensorcnt)
#define HAVE_TDM 1
#endif
#endif

// ---------------------------------------------------------------------------
// 1) fp32 -> fp16 elementwise convert (for x)
// ---------------------------------------------------------------------------
__global__ void mha_cvt_f16(const float* __restrict__ in,
                            _Float16* __restrict__ out, int n) {
    int i = blockIdx.x * blockDim.x + threadIdx.x;
    int stride = gridDim.x * blockDim.x;
    for (; i < n; i += stride) out[i] = (_Float16)in[i];
}

// ---------------------------------------------------------------------------
// 2) fp32 -> fp16 transpose convert for a 1024x1024 weight (WT[n][k] = W[k][n])
//    so that WMMA B-fragments become contiguous 32-byte loads.
// ---------------------------------------------------------------------------
__global__ __launch_bounds__(256)
void mha_cvt_wt(const float* __restrict__ W, _Float16* __restrict__ WT) {
    __shared__ float tile[32][33];
    const int x0 = blockIdx.x * 32, y0 = blockIdx.y * 32;
    const int tx = threadIdx.x & 31, ty = threadIdx.x >> 5;   // 32 x 8
    #pragma unroll
    for (int yy = ty; yy < 32; yy += 8)
        tile[yy][tx] = W[(size_t)(y0 + yy) * ND + (x0 + tx)];
    __syncthreads();
    #pragma unroll
    for (int yy = ty; yy < 32; yy += 8)
        WT[(size_t)(x0 + yy) * ND + (y0 + tx)] = (_Float16)tile[tx][yy];
}

// ---------------------------------------------------------------------------
// 3) Projection GEMM: Out = X(4096x1024,f16) * WT^T, written per-head.
//    transposed==0 : Out[b][h][s][dh]   (Q, K layouts)
//    transposed==1 : Out[b][h][dh][s]   (V^T layout, B-fragment-ready for P@V)
//    Block = 256 thr = 8 waves, block tile 128x128, wave tile 32x64.
//    Register double-buffered k-loop: fragments for k+32 are in flight while
//    the WMMAs for k issue, so waits are partial (not s_wait_loadcnt 0).
// ---------------------------------------------------------------------------
static __device__ __forceinline__
void gemm_load_a(AFrag a[2], const _Float16* __restrict__ X,
                 int wm, int ln, int hi, int k) {
    #pragma unroll
    for (int i = 0; i < 2; ++i) {
        const _Float16* ap = X + (size_t)(wm + i * 16 + ln) * ND + k;
        a[i].q[0] = *(const u32x4*)(ap + hi * 16);
        a[i].q[1] = *(const u32x4*)(ap + hi * 16 + 8);
    }
}
static __device__ __forceinline__
void gemm_load_b(AFrag b[4], const _Float16* __restrict__ WT,
                 int wn, int ln, int hi, int k) {
    #pragma unroll
    for (int j = 0; j < 4; ++j) {
        const _Float16* bp = WT + (size_t)(wn + j * 16 + ln) * ND + k + hi * 16;
        b[j].q[0] = ((const u32x4*)bp)[0];
        b[j].q[1] = ((const u32x4*)bp)[1];
    }
}
static __device__ __forceinline__
void gemm_mm(v8f acc[2][4], const AFrag a[2], const AFrag b[4]) {
    #pragma unroll
    for (int j = 0; j < 4; ++j)
        #pragma unroll
        for (int i = 0; i < 2; ++i)
            acc[i][j] = wmma_f16(a[i].v, b[j].v, acc[i][j]);
}

__global__ __launch_bounds__(256)
void mha_qkv_gemm(const _Float16* __restrict__ X,
                  const _Float16* __restrict__ WT,
                  _Float16* __restrict__ Out, int transposed) {
    const int lane = threadIdx.x & 31;
    const int wave = threadIdx.x >> 5;
    const int hi = lane >> 4, ln = lane & 15;

    const int wm = blockIdx.x * 128 + (wave & 3) * 32;   // gridDim.x = 32
    const int wn = blockIdx.y * 128 + (wave >> 2) * 64;  // gridDim.y = 8

    v8f acc[2][4] = {};
    AFrag a0[2], a1[2], b0[4], b1[4];
    gemm_load_a(a0, X, wm, ln, hi, 0);
    gemm_load_b(b0, WT, wn, ln, hi, 0);
    for (int k = 0; k < ND; k += 64) {
        gemm_load_a(a1, X, wm, ln, hi, k + 32);
        gemm_load_b(b1, WT, wn, ln, hi, k + 32);
        __builtin_prefetch(X + (size_t)(wm + ln) * ND + k + 64, 0, 1);
        gemm_mm(acc, a0, b0);
        if (k + 64 < ND) {
            gemm_load_a(a0, X, wm, ln, hi, k + 64);
            gemm_load_b(b0, WT, wn, ln, hi, k + 64);
        }
        gemm_mm(acc, a1, b1);
    }
    // C/D layout: lane holds col n=ln, rows r + hi*8
    #pragma unroll
    for (int i = 0; i < 2; ++i)
        #pragma unroll
        for (int j = 0; j < 4; ++j)
            #pragma unroll
            for (int r = 0; r < 8; ++r) {
                const int m = wm + i * 16 + r + hi * 8;   // b*2048 + s
                const int n = wn + j * 16 + ln;           // h*64 + dh
                const int b = m >> 11, s = m & (NS - 1);
                const int h = n >> 6, dh = n & (NDH - 1);
                const _Float16 val = (_Float16)acc[i][j][r];
                if (!transposed)
                    Out[((size_t)(b * NH + h) * NS + s) * NDH + dh] = val;
                else
                    Out[((size_t)(b * NH + h) * NDH + dh) * NS + s] = val;
            }
}

// ---------------------------------------------------------------------------
// 4) Flash attention. Block = 128 thr = 4 waves; block owns 64 q-rows of one
//    (b,h); each wave owns 16 q-rows. kv-tiles of 32 iterate to the BLOCK's
//    causal horizon (uniform trip count -> barriers legal); tiles fully
//    masked for a wave are numerically inert (p=0, alpha=1).
//    K tiles staged in LDS: TDM (tensor_load_to_lds) double-buffered one
//    tile ahead, completion via s_wait_tensorcnt; fallback cooperative fill.
// ---------------------------------------------------------------------------
#ifdef HAVE_TDM
static __device__ __forceinline__
void tdm_issue_k(const _Float16* gsrc, _Float16* ldst, int rows_left) {
    const unsigned lds_off = (unsigned)(uintptr_t)ldst;   // low 32b = LDS offset
    const unsigned long long ga = (unsigned long long)(uintptr_t)gsrc;
    v4u g0 = {};
    g0[0] = 1u;                                   // count=1, user descriptor
    g0[1] = lds_off;                              // lds_addr (bytes)
    g0[2] = (unsigned)(ga & 0xFFFFFFFFu);         // global_addr[31:0]
    g0[3] = (unsigned)((ga >> 32) & 0x1FFFFFFu)   // global_addr[56:32]
            | (2u << 30);                         // type = 2 ("image")
    v8i g1 = {};
    const unsigned dim0 = NDH;                    // tensor dim0 = 64 elements
    const unsigned dim1 = (unsigned)rows_left;    // remaining rows
    g1[0] = (int)(1u << 16);                      // data_size=1 (2B), no flags
    g1[1] = (int)((dim0 & 0xFFFFu) << 16);        // tensor_dim0[15:0]
    g1[2] = (int)(((dim0 >> 16) & 0xFFFFu) | ((dim1 & 0xFFFFu) << 16));
    g1[3] = (int)(((dim1 >> 16) & 0xFFFFu) | ((unsigned)NDH << 16)); // tile_dim0=64
    g1[4] = (int)32u;                             // tile_dim1=32, tile_dim2=0
    g1[5] = (int)NDH;                             // tensor_dim0_stride = 64
    g1[6] = 0;
    g1[7] = 0;
    v4i z4 = {};
    v8i z8 = {};
    // amdgpu-toolchain (clang-23) 6-arg form: (g0, g1, g2, g3, extra, cpol)
    __builtin_amdgcn_tensor_load_to_lds(g0, g1, z4, z4, z8, 0);
}
#endif

__global__ __launch_bounds__(128)
void mha_flash_attn(const _Float16* __restrict__ Qh,
                    const _Float16* __restrict__ Kh,
                    const _Float16* __restrict__ Vt,
                    float* __restrict__ Out) {
    const int lane = threadIdx.x & 31;
    const int wave = threadIdx.x >> 5;
    const int hi = lane >> 4, ln = lane & 15;
    const int b = blockIdx.z, h = blockIdx.y;
    const int qblk = blockIdx.x * 64;
    const int q0 = qblk + wave * 16;

    const _Float16* Q = Qh + (size_t)(b * NH + h) * NS * NDH;
    const _Float16* K = Kh + (size_t)(b * NH + h) * NS * NDH;
    const _Float16* V = Vt + (size_t)(b * NH + h) * NDH * NS;

    __shared__ _Float16 kbuf[2][32 * NDH];   // double-buffered K tile (2 x 4KB)
    __shared__ _Float16 pbuf[4][16 * 32];    // per-wave P staging (C->A relayout)
    _Float16* P = pbuf[wave];

    // Q A-fragments for dh-chunks 0..31 and 32..63 (loaded once)
    AFrag qf[2];
    {
        const _Float16* qp = Q + (size_t)(q0 + ln) * NDH;
        qf[0].q[0] = *(const u32x4*)(qp + hi * 16);
        qf[0].q[1] = *(const u32x4*)(qp + hi * 16 + 8);
        qf[1].q[0] = *(const u32x4*)(qp + 32 + hi * 16);
        qf[1].q[1] = *(const u32x4*)(qp + 40 + hi * 16);
    }

    v8f ctx[4] = {};
    float mrun[8], lrun[8];
    #pragma unroll
    for (int r = 0; r < 8; ++r) { mrun[r] = -3.0e38f; lrun[r] = 0.0f; }

    const float sc = 0.125f;                 // 1/sqrt(DH)
    const int ntiles = ((qblk + 63) >> 5) + 1;   // uniform across the block

#ifdef HAVE_TDM
    if (wave == 0) tdm_issue_k(K, kbuf[0], NS);
#endif

    int cur = 0;
    for (int t = 0; t < ntiles; ++t) {
        const int kv = t * 32;
#ifdef HAVE_TDM
        if (wave == 0) {
            if (t + 1 < ntiles) {
                tdm_issue_k(K + (size_t)(kv + 32) * NDH, kbuf[cur ^ 1],
                            NS - (kv + 32));
                __builtin_amdgcn_s_wait_tensorcnt(1);   // tile `cur` complete
            } else {
                __builtin_amdgcn_s_wait_tensorcnt(0);
            }
        }
#else
        {   // cooperative fill of kbuf[cur]: 32x64 f16 = 256 b128 chunks
            const u32x4* src = (const u32x4*)(K + (size_t)kv * NDH);
            u32x4* dst = (u32x4*)kbuf[cur];
            #pragma unroll
            for (int i = 0; i < 2; ++i)
                dst[threadIdx.x + i * 128] = src[threadIdx.x + i * 128];
        }
#endif
        __syncthreads();                     // K tile[cur] visible to all waves

        // V B-fragments for this kv tile (global; overlap with softmax math)
        AFrag vf[4];
        #pragma unroll
        for (int tt = 0; tt < 4; ++tt) {
            const _Float16* vp = V + (size_t)(tt * 16 + ln) * NS + kv + hi * 16;
            vf[tt].q[0] = ((const u32x4*)vp)[0];
            vf[tt].q[1] = ((const u32x4*)vp)[1];
        }

        // ---- scores S = Q*K^T for 16q x 32kv (K from LDS)
        v8f s[2] = {};
        #pragma unroll
        for (int cc = 0; cc < 2; ++cc) {
            #pragma unroll
            for (int c = 0; c < 2; ++c) {
                AFrag kf;
                const _Float16* kp =
                    kbuf[cur] + (size_t)(cc * 16 + ln) * NDH + c * 32 + hi * 16;
                kf.q[0] = ((const u32x4*)kp)[0];
                kf.q[1] = ((const u32x4*)kp)[1];
                s[cc] = wmma_f16(qf[c].v, kf.v, s[cc]);
            }
        }
        // ---- scale + causal mask + per-lane row max
        float tmax[8];
        #pragma unroll
        for (int r = 0; r < 8; ++r) tmax[r] = -3.0e38f;
        #pragma unroll
        for (int cc = 0; cc < 2; ++cc) {
            const int kvp = kv + cc * 16 + ln;
            #pragma unroll
            for (int r = 0; r < 8; ++r) {
                const int qq = q0 + r + hi * 8;
                float v = s[cc][r] * sc;
                v = (kvp <= qq) ? v : -3.0e38f;
                s[cc][r] = v;
                tmax[r] = fmaxf(tmax[r], v);
            }
        }
        // ---- row max across the 16-lane group (masks 1,2,4,8 stay in-group)
        #pragma unroll
        for (int r = 0; r < 8; ++r) {
            float v = tmax[r];
            v = fmaxf(v, __shfl_xor(v, 1, 32));
            v = fmaxf(v, __shfl_xor(v, 2, 32));
            v = fmaxf(v, __shfl_xor(v, 4, 32));
            v = fmaxf(v, __shfl_xor(v, 8, 32));
            tmax[r] = v;
        }
        float alpha[8], tsum[8];
        #pragma unroll
        for (int r = 0; r < 8; ++r) {
            const float mnew = fmaxf(mrun[r], tmax[r]);
            alpha[r] = __expf(mrun[r] - mnew);
            mrun[r] = mnew;
            tsum[r] = 0.0f;
        }
        // ---- p = exp(s - m), stage to per-wave LDS in [q][kv] order
        #pragma unroll
        for (int cc = 0; cc < 2; ++cc)
            #pragma unroll
            for (int r = 0; r < 8; ++r) {
                const float p = __expf(s[cc][r] - mrun[r]);
                tsum[r] += p;
                P[(r + hi * 8) * 32 + cc * 16 + ln] = (_Float16)p;
            }
        #pragma unroll
        for (int r = 0; r < 8; ++r) {
            float v = tsum[r];
            v += __shfl_xor(v, 1, 32);
            v += __shfl_xor(v, 2, 32);
            v += __shfl_xor(v, 4, 32);
            v += __shfl_xor(v, 8, 32);
            lrun[r] = lrun[r] * alpha[r] + v;
        }
        // ---- rescale accumulator
        #pragma unroll
        for (int tt = 0; tt < 4; ++tt)
            #pragma unroll
            for (int r = 0; r < 8; ++r) ctx[tt][r] *= alpha[r];

        // intra-wave cross-lane LDS visibility (P stores -> P fragment loads)
        asm volatile("s_wait_dscnt 0" ::: "memory");

        // ---- P as A-fragment (16q x 32kv), V^T as B-fragments, ctx += P@V
        AFrag pf;
        const _Float16* pp = P + ln * 32;
        pf.q[0] = *(const u32x4*)(pp + hi * 16);
        pf.q[1] = *(const u32x4*)(pp + hi * 16 + 8);
        #pragma unroll
        for (int tt = 0; tt < 4; ++tt)
            ctx[tt] = wmma_f16(pf.v, vf[tt].v, ctx[tt]);

        __syncthreads();                     // all reads of kbuf[cur] done
        cur ^= 1;
    }

    // ---- normalize and write fp32 output: out[b][s][h*64+dh]
    float inv[8];
    #pragma unroll
    for (int r = 0; r < 8; ++r) inv[r] = 1.0f / lrun[r];
    float* O = Out + (size_t)b * NS * ND + (size_t)h * NDH;
    #pragma unroll
    for (int tt = 0; tt < 4; ++tt)
        #pragma unroll
        for (int r = 0; r < 8; ++r) {
            const int qq = q0 + r + hi * 8;
            O[(size_t)qq * ND + tt * 16 + ln] = ctx[tt][r] * inv[r];
        }
}

// ---------------------------------------------------------------------------
// launch
// ---------------------------------------------------------------------------
extern "C" void kernel_launch(void* const* d_in, const int* in_sizes, int n_in,
                              void* d_out, int out_size, void* d_ws, size_t ws_size,
                              hipStream_t stream) {
    const float* x  = (const float*)d_in[0];
    const float* Wq = (const float*)d_in[1];
    const float* Wk = (const float*)d_in[2];
    const float* Wv = (const float*)d_in[3];
    float* out = (float*)d_out;

    char* ws = (char*)d_ws;
    const size_t MiB = 1u << 20;
    _Float16* xh  = (_Float16*)(ws + 0 * MiB);    // 8 MiB  x fp16
    _Float16* WqT = (_Float16*)(ws + 8 * MiB);    // 2 MiB
    _Float16* WkT = (_Float16*)(ws + 10 * MiB);   // 2 MiB
    _Float16* WvT = (_Float16*)(ws + 12 * MiB);   // 2 MiB
    _Float16* Qh  = (_Float16*)(ws + 14 * MiB);   // 8 MiB  [b][h][s][dh]
    _Float16* Kh  = (_Float16*)(ws + 22 * MiB);   // 8 MiB  [b][h][s][dh]
    _Float16* Vt  = (_Float16*)(ws + 30 * MiB);   // 8 MiB  [b][h][dh][s]

    const int nX = NB * NS * ND;                  // 4M elements
    mha_cvt_f16<<<4096, 256, 0, stream>>>(x, xh, nX);
    dim3 tgrid(ND / 32, ND / 32);
    mha_cvt_wt<<<tgrid, 256, 0, stream>>>(Wq, WqT);
    mha_cvt_wt<<<tgrid, 256, 0, stream>>>(Wk, WkT);
    mha_cvt_wt<<<tgrid, 256, 0, stream>>>(Wv, WvT);

    dim3 ggrid((NB * NS) / 128, ND / 128);        // 32 x 8
    mha_qkv_gemm<<<ggrid, 256, 0, stream>>>(xh, WqT, Qh, 0);
    mha_qkv_gemm<<<ggrid, 256, 0, stream>>>(xh, WkT, Kh, 0);
    mha_qkv_gemm<<<ggrid, 256, 0, stream>>>(xh, WvT, Vt, 1);

    dim3 agrid(NS / 64, NH, NB);                  // 32 x 16 x 2
    mha_flash_attn<<<agrid, 128, 0, stream>>>(Qh, Kh, Vt, out);
}
